// LocalnessMultiheadSelfAttention_24816321036396
// MI455X (gfx1250) — matrix operand order repaired and checked by reading, not compiled, not run on hardware
//
#include <hip/hip_runtime.h>

typedef __attribute__((ext_vector_type(16))) _Float16 v16h;
typedef __attribute__((ext_vector_type(8)))  float    v8f;

union V16H {
    v16h  v;
    uint4 q[2];
};

static __device__ __forceinline__ v8f vzero8() {
    v8f z = {0.f,0.f,0.f,0.f,0.f,0.f,0.f,0.f};
    return z;
}

#define WMMA_F16(a, b, c) \
    __builtin_amdgcn_wmma_f32_16x16x32_f16(false, (a), false, (b), (short)0, (c), false, false)

// Async global->LDS DMA (ASYNCcnt-tracked), gfx1250 VGLOBAL encoding.
// vdst = LDS byte offset (flat LDS addr truncates to addr[31:0], ISA 10.2).
static __device__ __forceinline__ void async_ld_b128(unsigned lds_off, const void* gptr) {
    asm volatile("global_load_async_to_lds_b128 %0, %1, off"
                 :: "v"(lds_off), "v"((unsigned long long)(size_t)gptr)
                 : "memory");
}
static __device__ __forceinline__ void wait_async0() {
    asm volatile("s_wait_asynccnt 0" ::: "memory");
}
static __device__ __forceinline__ unsigned lds_off_of(const void* p) {
    return (unsigned)(size_t)p;
}
// Native CDNA5 transcendental tanh. Inline asm bypasses the compiler's
// TRANS-op hazard tracking, so pad with v_nop (ISA 7.4: 1 indep op needed).
static __device__ __forceinline__ float tanh_hw(float x) {
    float r;
    asm("v_tanh_f32 %0, %1\n\tv_nop" : "=v"(r) : "v"(x));
    return r;
}

// ---------------------------------------------------------------------------
// Problem constants (from reference): T=2048, B=2, E=1024, H=16, hd=64
// ---------------------------------------------------------------------------
#define TT   2048
#define BB   2
#define EE   1024
#define HH   16
#define HD   64
#define BH   32      // B*H
#define MROWS 4096   // T*B
#define QSCALE 0.125f // hd^-0.5

// ---------------------------------------------------------------------------
// Kernel 0: f32 -> f16 conversion
// ---------------------------------------------------------------------------
__global__ void cvt_f16_kernel(const float* __restrict__ s, _Float16* __restrict__ d, int n) {
    int i = blockIdx.x * blockDim.x + threadIdx.x;
    if (i < n) d[i] = (_Float16)s[i];
}

// ---------------------------------------------------------------------------
// Kernel 1: fused QKV + localness projection GEMM.
//   C[4096, 4096] = X[4096,1024] @ [Wproj;Wloc]^T
// B tile (64n x 32k, 4KB) staged to LDS with async DMA, double buffered
// (runtime XOR'd buffer offset + unroll(1) keeps one loop body so the
//  accumulators stay in place -- no cross-body register shuffling).
// ---------------------------------------------------------------------------
__global__ __launch_bounds__(128) void qkv_loc_gemm_kernel(
    const _Float16* __restrict__ X,      // [4096,1024]
    const _Float16* __restrict__ Wproj,  // [3072,1024]
    const _Float16* __restrict__ Wloc,   // [1024,1024]
    const float*    __restrict__ Bias,   // [3072]
    _Float16* __restrict__ Qh,           // [32][2048][64]
    _Float16* __restrict__ Kh,           // [32][2048][64]
    _Float16* __restrict__ Vt,           // [32][64][2048]  (transposed)
    float*    __restrict__ Qloc)         // [32][2048][64]
{
    __shared__ __align__(16) _Float16 Bs[2][64 * 32];   // [n_local][k_local]
    const unsigned BUFSTRIDE = (unsigned)(sizeof(_Float16) * 64 * 32);  // 4KB

    const int tid  = threadIdx.x;
    const int wave = tid >> 5;
    const int lane = tid & 31;
    const int ln   = lane & 15;
    const int hi   = lane >> 4;
    const int m0   = blockIdx.x * 64 + wave * 16;
    const int n0   = blockIdx.y * 64;

    const _Float16* Wbase = (n0 < 3 * EE) ? (Wproj + (size_t)n0 * EE)
                                          : (Wloc + (size_t)(n0 - 3 * EE) * EE);

    // async stage: thread t copies row t/2, 32-byte half t%2 of the 64x32 tile
    const int srow  = tid >> 1;
    const int shalf = (tid & 1) * 16;   // in halves (16 halves == 32B)
    const unsigned lbase = lds_off_of(&Bs[0][srow * 32 + shalf]);
    const _Float16* gsrc = Wbase + (size_t)srow * EE + shalf;

    // prologue: stage k-block 0 into buffer 0
    async_ld_b128(lbase, gsrc);
    async_ld_b128(lbase + 16, gsrc + 8);

    v8f acc[4];
#pragma unroll
    for (int i = 0; i < 4; ++i) acc[i] = vzero8();

    unsigned cur = 0;
#pragma unroll 1
    for (int kb = 0; kb < EE / 32; ++kb) {
        wait_async0();
        __syncthreads();
        if (kb + 1 < EE / 32) {
            const unsigned l = lbase + (cur ^ BUFSTRIDE);
            const _Float16* g = gsrc + (kb + 1) * 32;
            async_ld_b128(l, g);
            async_ld_b128(l + 16, g + 8);
        }
        const _Float16* bs = (const _Float16*)((const char*)&Bs[0][0] + cur);
        cur ^= BUFSTRIDE;

        // A tile 16x32: lane<16 holds K {0..7,16..23}, lane>=16 {8..15,24..31}
        const _Float16* ap = X + (size_t)(m0 + ln) * EE + kb * 32 + hi * 8;
        V16H a;
        a.q[0] = *(const uint4*)(ap);
        a.q[1] = *(const uint4*)(ap + 16);
        __builtin_prefetch((const void*)(ap + 32), 0, 1);

#pragma unroll
        for (int nt = 0; nt < 4; ++nt) {
            // B tile 32x16: lane<16 holds K {0..7,16..23}, lane>=16 {8..15,24..31}
            const _Float16* wp = bs + (nt * 16 + ln) * 32 + hi * 8;
            V16H b;
            b.q[0] = *(const uint4*)(wp);
            b.q[1] = *(const uint4*)(wp + 16);
            acc[nt] = WMMA_F16(a.v, b.v, acc[nt]);
        }
    }

    // Epilogue: C layout -> lanes 0-15 rows 0..7, lanes 16-31 rows 8..15
#pragma unroll
    for (int nt = 0; nt < 4; ++nt) {
        const int n = n0 + nt * 16 + ln;
#pragma unroll
        for (int r = 0; r < 8; ++r) {
            const int m = m0 + r + hi * 8;  // row of X = t*B + b
            const int t = m >> 1;
            const int b = m & 1;
            const float v = acc[nt][r];
            if (n < 3 * EE) {
                const float val = v + Bias[n];
                const int s = n >> 10;        // 0=q 1=k 2=v
                const int e = n & (EE - 1);
                const int h = e >> 6;
                const int c = e & 63;
                const int bh = b * HH + h;
                if (s == 0)      Qh[((size_t)bh * TT + t) * HD + c] = (_Float16)(val * QSCALE);
                else if (s == 1) Kh[((size_t)bh * TT + t) * HD + c] = (_Float16)val;
                else             Vt[((size_t)bh * HD + c) * TT + t] = (_Float16)val;
            } else {
                const int e = n - 3 * EE;
                const int h = e >> 6;
                const int c = e & 63;
                const int bh = b * HH + h;
                Qloc[((size_t)bh * TT + t) * HD + c] = tanh_hw(v);
            }
        }
    }
}

// ---------------------------------------------------------------------------
// Kernel 2: localness center/width.  One wave per (head, t) row.
// ---------------------------------------------------------------------------
__global__ __launch_bounds__(128) void pd_kernel(
    const float* __restrict__ Qloc,  // [32][2048][64]
    const float* __restrict__ uP,    // [64]
    const float* __restrict__ uD,    // [64]
    float* __restrict__ Pc,          // [32*2048]
    float* __restrict__ Ic)          // [32*2048]
{
    const int wave = threadIdx.x >> 5;
    const int lane = threadIdx.x & 31;
    const int row  = blockIdx.x * 4 + wave;   // 0 .. 32*2048-1

    const float* q = Qloc + (size_t)row * HD;
    const float x0 = q[lane];
    const float x1 = q[lane + 32];
    float sp = x0 * uP[lane] + x1 * uP[lane + 32];
    float sd = x0 * uD[lane] + x1 * uD[lane + 32];
#pragma unroll
    for (int off = 16; off >= 1; off >>= 1) {
        sp += __shfl_xor(sp, off, 32);
        sd += __shfl_xor(sd, off, 32);
    }
    if (lane == 0) {
        const float p = (float)TT / (1.f + __expf(-sp));
        const float d = (float)TT / (1.f + __expf(-sd));
        Pc[row] = p;
        Ic[row] = 2.f / (d * d);
    }
}

// ---------------------------------------------------------------------------
// Kernel 3: flash attention with Gaussian localness bias.
// Grid: 32 heads * 32 query-blocks(64). Block: 128 threads (4 waves).
// K tile (32x64) and V tile (64x32) are shared by all 4 waves -> staged once
// per block into double-buffered LDS via async DMA, overlapped with compute.
// ---------------------------------------------------------------------------
__global__ __launch_bounds__(128) void attn_kernel(
    const _Float16* __restrict__ Qh,   // [32][2048][64] (pre-scaled)
    const _Float16* __restrict__ Kh,   // [32][2048][64]
    const _Float16* __restrict__ Vt,   // [32][64][2048]
    const float*    __restrict__ Pc,   // [32*2048]
    const float*    __restrict__ Ic,   // [32*2048]
    _Float16* __restrict__ AttnOut)    // [4096][1024]
{
    __shared__ __align__(16) _Float16 Ks[2][32 * 64];   // [j_local][hd]
    __shared__ __align__(16) _Float16 Vs[2][64 * 32];   // [hd][j_local]
    __shared__ __align__(16) _Float16 plds[4][16 * 32];
    const unsigned BUFSTRIDE = (unsigned)(sizeof(_Float16) * 32 * 64);  // 4KB

    const int tid  = threadIdx.x;
    const int wave = tid >> 5;
    const int lane = tid & 31;
    const int ln   = lane & 15;
    const int hi   = lane >> 4;
    const int bh   = blockIdx.x >> 5;
    const int qb   = blockIdx.x & 31;
    const int m0   = qb * 64 + wave * 16;   // query row base (t within head)

    // Per-thread staging slots: K row = tid/4 (128B rows), V row = tid/2 (64B)
    const unsigned kl0 = lds_off_of(&Ks[0][(tid >> 2) * 64 + (tid & 3) * 16]);
    const unsigned vl0 = lds_off_of(&Vs[0][(tid >> 1) * 32 + (tid & 1) * 16]);
    const _Float16* kg0 = Kh + ((size_t)bh * TT + (tid >> 2)) * HD + (tid & 3) * 16;
    const _Float16* vg0 = Vt + ((size_t)bh * HD + (tid >> 1)) * TT + (tid & 1) * 16;

    // prologue: stage key-block 0 into buffer 0
    async_ld_b128(kl0,      kg0);
    async_ld_b128(kl0 + 16, kg0 + 8);
    async_ld_b128(vl0,      vg0);
    async_ld_b128(vl0 + 16, vg0 + 8);

    // Q as two A-tiles (hd 0..31 and 32..63), register-resident all loop long
    const _Float16* qp = Qh + ((size_t)bh * TT + (m0 + ln)) * HD + hi * 8;
    V16H aQ0, aQ1;
    aQ0.q[0] = *(const uint4*)(qp);
    aQ0.q[1] = *(const uint4*)(qp + 16);
    aQ1.q[0] = *(const uint4*)(qp + 32);
    aQ1.q[1] = *(const uint4*)(qp + 48);

    const int rowb = m0 + hi * 8;           // first of this lane group's 8 rows
    float pr[8], ir[8];
#pragma unroll
    for (int r = 0; r < 8; ++r) {
        pr[r] = Pc[bh * TT + rowb + r];
        ir[r] = Ic[bh * TT + rowb + r];
    }

    float mrun[8], lrun[8];
#pragma unroll
    for (int r = 0; r < 8; ++r) { mrun[r] = -1e30f; lrun[r] = 0.f; }
    v8f o[4];
#pragma unroll
    for (int i = 0; i < 4; ++i) o[i] = vzero8();

    _Float16* pl = plds[wave];

    unsigned cur = 0;
#pragma unroll 1
    for (int j0 = 0; j0 < TT; j0 += 32) {
        wait_async0();
        __syncthreads();
        if (j0 + 32 < TT) {
            const unsigned nb = cur ^ BUFSTRIDE;
            async_ld_b128(kl0 + nb,      kg0 + (size_t)(j0 + 32) * HD);
            async_ld_b128(kl0 + nb + 16, kg0 + (size_t)(j0 + 32) * HD + 8);
            async_ld_b128(vl0 + nb,      vg0 + j0 + 32);
            async_ld_b128(vl0 + nb + 16, vg0 + j0 + 40);
        }
        const _Float16* ks = (const _Float16*)((const char*)&Ks[0][0] + cur);
        const _Float16* vs = (const _Float16*)((const char*)&Vs[0][0] + cur);
        cur ^= BUFSTRIDE;

        // ---- S = Q @ K^T for 32 keys (two 16x16 tiles), K from LDS ----
        v8f s0 = vzero8(), s1 = vzero8();
        {
            const _Float16* kp0 = ks + ln * 64 + hi * 8;
            V16H b0, b1;
            b0.q[0] = *(const uint4*)(kp0);
            b0.q[1] = *(const uint4*)(kp0 + 16);
            b1.q[0] = *(const uint4*)(kp0 + 32);
            b1.q[1] = *(const uint4*)(kp0 + 48);
            s0 = WMMA_F16(aQ0.v, b0.v, s0);
            s0 = WMMA_F16(aQ1.v, b1.v, s0);
            const _Float16* kp1 = ks + (16 + ln) * 64 + hi * 8;
            b0.q[0] = *(const uint4*)(kp1);
            b0.q[1] = *(const uint4*)(kp1 + 16);
            b1.q[0] = *(const uint4*)(kp1 + 32);
            b1.q[1] = *(const uint4*)(kp1 + 48);
            s1 = WMMA_F16(aQ0.v, b0.v, s1);
            s1 = WMMA_F16(aQ1.v, b1.v, s1);
        }

        // ---- localness bias + online softmax update ----
        const float jf = (float)(j0 + ln);
        float p0r[8], p1r[8], corr[8];
#pragma unroll
        for (int r = 0; r < 8; ++r) {
            const float d0 = jf - pr[r];
            const float d1 = jf + 16.f - pr[r];
            const float x0 = s0[r] - d0 * d0 * ir[r];
            const float x1 = s1[r] - d1 * d1 * ir[r];
            float tm = fmaxf(x0, x1);
            tm = fmaxf(tm, __shfl_xor(tm, 1, 32));
            tm = fmaxf(tm, __shfl_xor(tm, 2, 32));
            tm = fmaxf(tm, __shfl_xor(tm, 4, 32));
            tm = fmaxf(tm, __shfl_xor(tm, 8, 32));
            const float nm = fmaxf(mrun[r], tm);
            const float c  = __expf(mrun[r] - nm);
            const float e0 = __expf(x0 - nm);
            const float e1 = __expf(x1 - nm);
            float rs = e0 + e1;
            rs += __shfl_xor(rs, 1, 32);
            rs += __shfl_xor(rs, 2, 32);
            rs += __shfl_xor(rs, 4, 32);
            rs += __shfl_xor(rs, 8, 32);
            lrun[r] = lrun[r] * c + rs;
            mrun[r] = nm;
            corr[r] = c;
            p0r[r] = e0;
            p1r[r] = e1;
        }
#pragma unroll
        for (int nt = 0; nt < 4; ++nt)
#pragma unroll
            for (int r = 0; r < 8; ++r)
                o[nt][r] *= corr[r];

        // ---- transpose P (C layout -> A layout) through per-wave LDS ----
#pragma unroll
        for (int r = 0; r < 8; ++r) {
            pl[(r + hi * 8) * 32 + ln]      = (_Float16)p0r[r];
            pl[(r + hi * 8) * 32 + ln + 16] = (_Float16)p1r[r];
        }
        asm volatile("s_wait_dscnt 0" ::: "memory");  // wave-local LDS RAW fence
        V16H aP;
        aP.q[0] = *(const uint4*)&pl[ln * 32 + hi * 8];
        aP.q[1] = *(const uint4*)&pl[ln * 32 + hi * 8 + 16];

        // ---- O += P @ V, V from LDS ----
#pragma unroll
        for (int nt = 0; nt < 4; ++nt) {
            const _Float16* vp = vs + (nt * 16 + ln) * 32 + hi * 8;
            V16H bV;
            bV.q[0] = *(const uint4*)(vp);
            bV.q[1] = *(const uint4*)(vp + 16);
            o[nt] = WMMA_F16(aP.v, bV.v, o[nt]);
        }
    }

    // ---- normalize and scatter back to [t*B+b][h*64+c] f16 ----
    const int b = bh >> 4;
    const int h = bh & 15;
    float linv[8];
#pragma unroll
    for (int r = 0; r < 8; ++r) linv[r] = 1.f / lrun[r];
#pragma unroll
    for (int nt = 0; nt < 4; ++nt) {
        const int col = h * HD + nt * 16 + ln;
#pragma unroll
        for (int r = 0; r < 8; ++r) {
            const int t = m0 + r + hi * 8;
            AttnOut[(size_t)(t * BB + b) * EE + col] = (_Float16)(o[nt][r] * linv[r]);
        }
    }
}

// ---------------------------------------------------------------------------
// Kernel 4: output projection GEMM + bias -> d_out f32.
// ---------------------------------------------------------------------------
__global__ __launch_bounds__(128) void out_gemm_kernel(
    const _Float16* __restrict__ A,    // [4096,1024]
    const _Float16* __restrict__ W,    // [1024,1024]
    const float*    __restrict__ Bias, // [1024]
    float* __restrict__ Out)           // [4096,1024]
{
    __shared__ __align__(16) _Float16 Bs[2][64 * 32];
    const unsigned BUFSTRIDE = (unsigned)(sizeof(_Float16) * 64 * 32);

    const int tid  = threadIdx.x;
    const int wave = tid >> 5;
    const int lane = tid & 31;
    const int ln   = lane & 15;
    const int hi   = lane >> 4;
    const int m0   = blockIdx.x * 64 + wave * 16;
    const int n0   = blockIdx.y * 64;

    const int srow  = tid >> 1;
    const int shalf = (tid & 1) * 16;
    const unsigned lbase = lds_off_of(&Bs[0][srow * 32 + shalf]);
    const _Float16* gsrc = W + (size_t)(n0 + srow) * EE + shalf;

    async_ld_b128(lbase, gsrc);
    async_ld_b128(lbase + 16, gsrc + 8);

    v8f acc[4];
#pragma unroll
    for (int i = 0; i < 4; ++i) acc[i] = vzero8();

    unsigned cur = 0;
#pragma unroll 1
    for (int kb = 0; kb < EE / 32; ++kb) {
        wait_async0();
        __syncthreads();
        if (kb + 1 < EE / 32) {
            const unsigned l = lbase + (cur ^ BUFSTRIDE);
            const _Float16* g = gsrc + (kb + 1) * 32;
            async_ld_b128(l, g);
            async_ld_b128(l + 16, g + 8);
        }
        const _Float16* bs = (const _Float16*)((const char*)&Bs[0][0] + cur);
        cur ^= BUFSTRIDE;

        const _Float16* ap = A + (size_t)(m0 + ln) * EE + kb * 32 + hi * 8;
        V16H a;
        a.q[0] = *(const uint4*)(ap);
        a.q[1] = *(const uint4*)(ap + 16);
        __builtin_prefetch((const void*)(ap + 32), 0, 1);

#pragma unroll
        for (int nt = 0; nt < 4; ++nt) {
            const _Float16* wp = bs + (nt * 16 + ln) * 32 + hi * 8;
            V16H b;
            b.q[0] = *(const uint4*)(wp);
            b.q[1] = *(const uint4*)(wp + 16);
            acc[nt] = WMMA_F16(a.v, b.v, acc[nt]);
        }
    }

#pragma unroll
    for (int nt = 0; nt < 4; ++nt) {
        const int n = n0 + nt * 16 + ln;
        const float bs = Bias[n];
#pragma unroll
        for (int r = 0; r < 8; ++r) {
            const int m = m0 + r + hi * 8;
            Out[(size_t)m * EE + n] = acc[nt][r] + bs;
        }
    }
}

// ---------------------------------------------------------------------------
// Host-side orchestration
// ---------------------------------------------------------------------------
extern "C" void kernel_launch(void* const* d_in, const int* in_sizes, int n_in,
                              void* d_out, int out_size, void* d_ws, size_t ws_size,
                              hipStream_t stream) {
    const float* query     = (const float*)d_in[0];   // [2048,2,1024]
    const float* in_proj_w = (const float*)d_in[1];   // [3072,1024]
    const float* in_proj_b = (const float*)d_in[2];   // [3072]
    const float* loc_w     = (const float*)d_in[3];   // [1024,1024]
    const float* uP        = (const float*)d_in[4];   // [64]
    const float* uD        = (const float*)d_in[5];   // [64]
    const float* out_w     = (const float*)d_in[6];   // [1024,1024]
    const float* out_b     = (const float*)d_in[7];   // [1024]
    float* out = (float*)d_out;

    char* ws = (char*)d_ws;
    size_t off = 0;
    auto alloc = [&](size_t bytes) { char* p = ws + off; off = (off + bytes + 255) & ~(size_t)255; return p; };

    _Float16* Xf16   = (_Float16*)alloc((size_t)MROWS * EE * 2);       // 8 MB
    _Float16* Wp16   = (_Float16*)alloc((size_t)3 * EE * EE * 2);      // 6 MB
    _Float16* Wl16   = (_Float16*)alloc((size_t)EE * EE * 2);          // 2 MB
    _Float16* Wo16   = (_Float16*)alloc((size_t)EE * EE * 2);          // 2 MB
    _Float16* Qh     = (_Float16*)alloc((size_t)BH * TT * HD * 2);     // 8 MB
    _Float16* Kh     = (_Float16*)alloc((size_t)BH * TT * HD * 2);     // 8 MB
    _Float16* Vt     = (_Float16*)alloc((size_t)BH * HD * TT * 2);     // 8 MB
    float*    Qloc   = (float*)   alloc((size_t)BH * TT * HD * 4);     // 16 MB
    float*    Pc     = (float*)   alloc((size_t)BH * TT * 4);
    float*    Ic     = (float*)   alloc((size_t)BH * TT * 4);
    _Float16* AttnO  = (_Float16*)alloc((size_t)MROWS * EE * 2);       // 8 MB
    (void)ws_size; (void)n_in; (void)in_sizes; (void)out_size;

    // 0) weight / activation conversion to f16
    {
        int n;
        n = MROWS * EE;     cvt_f16_kernel<<<(n + 255) / 256, 256, 0, stream>>>(query, Xf16, n);
        n = 3 * EE * EE;    cvt_f16_kernel<<<(n + 255) / 256, 256, 0, stream>>>(in_proj_w, Wp16, n);
        n = EE * EE;        cvt_f16_kernel<<<(n + 255) / 256, 256, 0, stream>>>(loc_w, Wl16, n);
        n = EE * EE;        cvt_f16_kernel<<<(n + 255) / 256, 256, 0, stream>>>(out_w, Wo16, n);
    }

    // 1) fused QKV + loc projection
    qkv_loc_gemm_kernel<<<dim3(MROWS / 64, 4 * EE / 64), 128, 0, stream>>>(
        Xf16, Wp16, Wl16, in_proj_b, Qh, Kh, Vt, Qloc);

    // 2) localness p / inverse-width
    pd_kernel<<<(BH * TT) / 4, 128, 0, stream>>>(Qloc, uP, uD, Pc, Ic);

    // 3) flash attention with localness bias
    attn_kernel<<<BH * (TT / 64), 128, 0, stream>>>(Qh, Kh, Vt, Pc, Ic, AttnO);

    // 4) output projection
    out_gemm_kernel<<<dim3(MROWS / 64, EE / 64), 128, 0, stream>>>(AttnO, Wo16, out_b, out);
}
